// Model_83141976916202
// MI455X (gfx1250) — matrix operand order repaired and checked by reading, not compiled, not run on hardware
//
#include <hip/hip_runtime.h>

#define T_DIM 1024
#define B_DIM 256
#define I_DIM 128
#define H_DIM 256
#define L_DIM 6
#define G3H   768   // 3*H

typedef __attribute__((ext_vector_type(16))) __bf16 v16bf;
typedef __attribute__((ext_vector_type(8)))  float  v8f;
typedef unsigned short u16;
typedef unsigned int   u32;
typedef unsigned long long u64;

union Frag16 {          // 16 bf16 elements = one WMMA A or B fragment (8 VGPRs)
    v16bf v;
    u32   q[8];
};

__device__ __forceinline__ u16 f2bf(float f) {
    u32 u = __builtin_bit_cast(u32, f);
    u32 r = u + 0x7FFFu + ((u >> 16) & 1u);   // round-to-nearest-even
    return (u16)(r >> 16);
}

__device__ __forceinline__ float sigmoid_fast(float x) {
    return 1.0f / (1.0f + __expf(-x));
}

__device__ __forceinline__ v8f v8f_zero() {
    v8f z = {0.f, 0.f, 0.f, 0.f, 0.f, 0.f, 0.f, 0.f};
    return z;
}

// Low 32 bits of a generic LDS address == byte offset in LDS
// (ISA: aperture decode).
__device__ __forceinline__ u32 lds_offset(const void* p) {
    return (u32)(u64)p;
}

// ---------------------------------------------------------------------------
// fp32 -> bf16 conversion (grid-strided)
// ---------------------------------------------------------------------------
__global__ void k_f32_to_bf16(const float* __restrict__ src, u16* __restrict__ dst, int n) {
    int i = blockIdx.x * blockDim.x + threadIdx.x;
    int stride = gridDim.x * blockDim.x;
    for (; i < n; i += stride) dst[i] = f2bf(src[i]);
}

// ---------------------------------------------------------------------------
// Input projection: xg[M,768] = A[M,K](bf16) @ W[768,K]^T(bf16) + bias (fp32)
// grid = (M/128, 768/128); 256 threads = 8 waves; wave tile = 32M x 64N.
// ---------------------------------------------------------------------------
__global__ void k_proj(const u16* __restrict__ A, const u16* __restrict__ W,
                       const float* __restrict__ bias, float* __restrict__ xg, int K) {
    const int lane = threadIdx.x & 31;
    const int wave = threadIdx.x >> 5;
    const int half = lane >> 4;
    const int l15  = lane & 15;
    const int m0 = blockIdx.x * 128 + (wave & 3) * 32;
    const int n0 = blockIdx.y * 128 + (wave >> 2) * 64;

    v8f acc[2][4];
#pragma unroll
    for (int mi = 0; mi < 2; ++mi)
#pragma unroll
        for (int ni = 0; ni < 4; ++ni) acc[mi][ni] = v8f_zero();

    for (int k0 = 0; k0 < K; k0 += 32) {
        Frag16 a[2], b[4];
#pragma unroll
        for (int mi = 0; mi < 2; ++mi) {
            const u16* p = A + (size_t)(m0 + mi * 16 + l15) * K + k0 + 8 * half;
            *(uint4*)&a[mi].q[0] = *(const uint4*)p;
            *(uint4*)&a[mi].q[4] = *(const uint4*)(p + 16);
        }
#pragma unroll
        for (int ni = 0; ni < 4; ++ni) {
            const u16* p = W + (size_t)(n0 + ni * 16 + l15) * K + k0 + 8 * half;
            *(uint4*)&b[ni].q[0] = *(const uint4*)p;
            *(uint4*)&b[ni].q[4] = *(const uint4*)(p + 16);
        }
#pragma unroll
        for (int mi = 0; mi < 2; ++mi)
#pragma unroll
            for (int ni = 0; ni < 4; ++ni)
                acc[mi][ni] = __builtin_amdgcn_wmma_f32_16x16x32_bf16(
                    false, a[mi].v, false, b[ni].v, (short)0, acc[mi][ni], false, false);
    }

#pragma unroll
    for (int mi = 0; mi < 2; ++mi)
#pragma unroll
        for (int ni = 0; ni < 4; ++ni) {
            int n = n0 + ni * 16 + l15;
            float bv = bias[n];
#pragma unroll
            for (int r = 0; r < 8; ++r) {
                int m = m0 + mi * 16 + r + 8 * half;
                xg[(size_t)m * G3H + n] = acc[mi][ni][r] + bv;
            }
        }
}

// ---------------------------------------------------------------------------
// Recurrent scan. One WG per 16 batch rows (grid = B/16 = 16, block = 256).
//  * Whh slice held in REGISTERS per wave (persistent-RNN): 48 fragments,
//    384 VGPRs, loaded once before the time loop. Inner loop = LDS A-frags
//    + 48 v_wmma, no global weight traffic on the recurrent critical path.
//  * xg[t] slice (16 x 768 fp32, 48 KB) staged into LDS with
//    global_load_async_to_lds_b128, issued before the GEMM so the fetch
//    overlaps the matrix work; s_wait_asynccnt 0 + barrier before gate math.
//  * Each wave owns all THREE gates for its 32 hidden columns, so r/z/n land
//    in identical C-fragment slots and the gate math never leaves registers.
// ---------------------------------------------------------------------------
__global__ void k_scan(const float* __restrict__ xg,    // [T*B, 768] incl. b_ih
                       const u16*  __restrict__ Whh,    // [768, 256] bf16
                       const float* __restrict__ bhh,   // [768]
                       const float* __restrict__ h0l,   // [B, 256]
                       u16*  __restrict__ out_bf,       // [T*B, 256] or null
                       float* __restrict__ out_f)       // [T*B, 256] or null
{
    __shared__ u16   h_bf[16][H_DIM];   //  8 KB
    __shared__ float h_f [16][H_DIM];   // 16 KB
    __shared__ float xbuf[16][G3H];     // 48 KB async-staged xg slice

    const int lane  = threadIdx.x & 31;
    const int wave  = threadIdx.x >> 5;
    const int half  = lane >> 4;
    const int l15   = lane & 15;
    const int bbase = blockIdx.x * 16;

    // ---- preload this wave's Whh slice into registers (once) ----
    Frag16 wfrag[3][2][8];
#pragma unroll
    for (int g = 0; g < 3; ++g)
#pragma unroll
        for (int jt = 0; jt < 2; ++jt)
#pragma unroll
            for (int kk = 0; kk < 8; ++kk) {
                int n = g * H_DIM + wave * 32 + jt * 16 + l15;
                const u16* p = Whh + (size_t)n * H_DIM + kk * 32 + 8 * half;
                *(uint4*)&wfrag[g][jt][kk].q[0] = *(const uint4*)p;
                *(uint4*)&wfrag[g][jt][kk].q[4] = *(const uint4*)(p + 16);
            }

    // ---- init h ----
    for (int e = threadIdx.x; e < 16 * H_DIM; e += blockDim.x) {
        int b = e >> 8, j = e & (H_DIM - 1);
        float v = h0l[(size_t)(bbase + b) * H_DIM + j];
        h_f[b][j]  = v;
        h_bf[b][j] = f2bf(v);
    }
    __syncthreads();

    float bias_g[3][2];
#pragma unroll
    for (int g = 0; g < 3; ++g)
#pragma unroll
        for (int jt = 0; jt < 2; ++jt)
            bias_g[g][jt] = bhh[g * H_DIM + wave * 32 + jt * 16 + l15];

    const u32 xbuf_base = lds_offset(&xbuf[0][0]);

    for (int t = 0; t < T_DIM; ++t) {
        // ---- issue async stage of xg[t] slice (overlaps the GEMM below) ----
        {
            const float* src = xg + (size_t)(t * B_DIM + bbase) * G3H;  // 48 KB contiguous
#pragma unroll
            for (int c = 0; c < 12; ++c) {
                int chunk = threadIdx.x + c * 256;            // 0..3071 x 16B
                u64 ga = (u64)(src + (size_t)chunk * 4);
                u32 la = xbuf_base + (u32)chunk * 16u;
                asm volatile("global_load_async_to_lds_b128 %0, %1, off"
                             :: "v"(la), "v"(ga) : "memory");
            }
        }

        // ---- recurrent GEMM from register-resident weights ----
        v8f acc[3][2];
#pragma unroll
        for (int g = 0; g < 3; ++g)
#pragma unroll
            for (int jt = 0; jt < 2; ++jt) acc[g][jt] = v8f_zero();

#pragma unroll
        for (int kk = 0; kk < 8; ++kk) {
            Frag16 a;
            {
                const u16* p = &h_bf[l15][kk * 32 + 8 * half];
                *(uint4*)&a.q[0] = *(const uint4*)p;
                *(uint4*)&a.q[4] = *(const uint4*)(p + 16);
            }
#pragma unroll
            for (int g = 0; g < 3; ++g)
#pragma unroll
                for (int jt = 0; jt < 2; ++jt)
                    acc[g][jt] = __builtin_amdgcn_wmma_f32_16x16x32_bf16(
                        false, a.v, false, wfrag[g][jt][kk].v, (short)0,
                        acc[g][jt], false, false);
        }

        // async xg slice complete + everyone done reading h_bf
        asm volatile("s_wait_asynccnt 0" ::: "memory");
        __syncthreads();

        // ---- fused gate math (registers + LDS only) ----
#pragma unroll
        for (int jt = 0; jt < 2; ++jt) {
            int j = wave * 32 + jt * 16 + l15;
#pragma unroll
            for (int r = 0; r < 8; ++r) {
                int m = r + 8 * half;                 // batch row in slice
                float hr = acc[0][jt][r] + bias_g[0][jt];
                float hz = acc[1][jt][r] + bias_g[1][jt];
                float hn = acc[2][jt][r] + bias_g[2][jt];
                float rg = sigmoid_fast(xbuf[m][j] + hr);
                float zg = sigmoid_fast(xbuf[m][H_DIM + j] + hz);
                float ng = tanhf(xbuf[m][2 * H_DIM + j] + rg * hn);
                float hprev = h_f[m][j];
                float hnew  = (1.0f - zg) * ng + zg * hprev;
                h_f[m][j]  = hnew;
                h_bf[m][j] = f2bf(hnew);
                size_t oidx = (size_t)(t * B_DIM + bbase + m) * H_DIM + j;
                if (out_f) out_f[oidx] = hnew;
                else       out_bf[oidx] = f2bf(hnew);
            }
        }
        __syncthreads();   // h updated before next step's GEMM
    }
}

// ---------------------------------------------------------------------------
extern "C" void kernel_launch(void* const* d_in, const int* in_sizes, int n_in,
                              void* d_out, int out_size, void* d_ws, size_t ws_size,
                              hipStream_t stream) {
    const float* x     = (const float*)d_in[0];   // [T,B,I]
    const float* h0    = (const float*)d_in[1];   // [L,B,H]
    const float* w_ih0 = (const float*)d_in[2];   // [3H,I]
    const float* w_ih  = (const float*)d_in[3];   // [L-1,3H,H]
    const float* w_hh  = (const float*)d_in[4];   // [L,3H,H]
    const float* b_ih  = (const float*)d_in[5];   // [L,3H]
    const float* b_hh  = (const float*)d_in[6];   // [L,3H]
    float* out = (float*)d_out;                   // [T,B,H]

    const size_t M = (size_t)T_DIM * B_DIM;       // 262144 rows

    char* ws = (char*)d_ws;
    u16* bufA = (u16*)ws;  ws += M * H_DIM * sizeof(u16);                       // 128 MB
    u16* bufB = (u16*)ws;  ws += M * H_DIM * sizeof(u16);                       // 128 MB
    float* xg = (float*)ws; ws += M * G3H * sizeof(float);                      // 768 MB
    u16* wih0_bf = (u16*)ws; ws += (size_t)G3H * I_DIM * sizeof(u16);
    u16* wih_bf  = (u16*)ws; ws += (size_t)(L_DIM - 1) * G3H * H_DIM * sizeof(u16);
    u16* whh_bf  = (u16*)ws; ws += (size_t)L_DIM * G3H * H_DIM * sizeof(u16);

    // weight + input conversions (deterministic each call)
    k_f32_to_bf16<<<128,  256, 0, stream>>>(w_ih0, wih0_bf, G3H * I_DIM);
    k_f32_to_bf16<<<512,  256, 0, stream>>>(w_ih,  wih_bf,  (L_DIM - 1) * G3H * H_DIM);
    k_f32_to_bf16<<<512,  256, 0, stream>>>(w_hh,  whh_bf,  L_DIM * G3H * H_DIM);
    k_f32_to_bf16<<<2048, 256, 0, stream>>>(x,     bufA,    (int)(M * I_DIM));

    u16* cur = bufA;
    u16* nxt = bufB;
    for (int l = 0; l < L_DIM; ++l) {
        const int K = (l == 0) ? I_DIM : H_DIM;
        const u16* Wih = (l == 0) ? wih0_bf : wih_bf + (size_t)(l - 1) * G3H * H_DIM;

        dim3 pgrid((unsigned)(M / 128), G3H / 128);
        k_proj<<<pgrid, 256, 0, stream>>>(cur, Wih, b_ih + (size_t)l * G3H, xg, K);

        const bool last = (l == L_DIM - 1);
        k_scan<<<B_DIM / 16, 256, 0, stream>>>(
            xg, whh_bf + (size_t)l * G3H * H_DIM, b_hh + (size_t)l * G3H,
            h0 + (size_t)l * B_DIM * H_DIM,
            last ? nullptr : nxt, last ? out : nullptr);

        u16* tmp = cur; cur = nxt; nxt = tmp;
    }
}
